// RelationalGraphConvolution_14027363188776
// MI455X (gfx1250) — matrix operand order repaired and checked by reading, not compiled, not run on hardware
//
#include <hip/hip_runtime.h>
#include <hip/hip_bf16.h>

#define B_    16
#define N_    512
#define R_    10
#define DE_   256
#define DR_   64
#define DOUT_ 256
#define INDIM_ 3200   // (DE+DR)*R

typedef __attribute__((ext_vector_type(16))) __bf16 v16bf;
typedef __attribute__((ext_vector_type(8)))  float  v8f;

__device__ __forceinline__ unsigned short f2bf(float f) {
    unsigned int u = __float_as_uint(f);
    u += 0x7fffu + ((u >> 16) & 1u);   // round-to-nearest-even
    return (unsigned short)(u >> 16);
}

union ABf16 { v16bf v; uint4 q[2]; unsigned short s[16]; };

// ---------------------------------------------------------------------------
// Prep kernels: bf16 conversions + tiny relation-weight dot products
// ---------------------------------------------------------------------------
__global__ void k_cvt_node(const float* __restrict__ x,
                           unsigned short* __restrict__ y, int n) {
    int i = blockIdx.x * blockDim.x + threadIdx.x;
    if (i < n) y[i] = f2bf(x[i]);
}

// Wbf[r][dout][d] = bf16(W[dout, r*(DE+DR)+d]), d < DE   (K-contiguous B operand)
__global__ void k_cvt_w(const float* __restrict__ W,
                        unsigned short* __restrict__ Wbf) {
    int i = blockIdx.x * blockDim.x + threadIdx.x;
    if (i >= R_ * DOUT_ * DE_) return;
    int d = i % DE_;
    int t = i / DE_;
    int dout = t % DOUT_;
    int r = t / DOUT_;
    Wbf[i] = f2bf(W[dout * INDIM_ + r * (DE_ + DR_) + d]);
}

// S[b,r,dout] = rel[b,r,:] . W[dout, r*320+256 : r*320+320]   (f32, exact)
__global__ void k_srel(const float* __restrict__ rel,
                       const float* __restrict__ W,
                       float* __restrict__ S) {
    int i = blockIdx.x * blockDim.x + threadIdx.x;
    if (i >= B_ * R_ * DOUT_) return;
    int dout = i % DOUT_;
    int t = i / DOUT_;
    int r = t % R_;
    int b = t / R_;
    const float* rp = rel + (b * R_ + r) * DR_;
    const float* wp = W + dout * INDIM_ + r * (DE_ + DR_) + DE_;
    float s = 0.f;
#pragma unroll
    for (int d = 0; d < DR_; ++d) s += rp[d] * wp[d];
    S[i] = s;
}

// ---------------------------------------------------------------------------
// Kernel 1: Y[b,r][dout,m] = bf16( node[b,m,:DE] . Wn_r[dout,:] )   (K-major!)
// One wave (32 threads) per 16x16 tile; 8x v_wmma_f32_16x16x32_bf16.
// ---------------------------------------------------------------------------
__global__ __launch_bounds__(32)
void k_ygemm(const unsigned short* __restrict__ nodebf,
             const unsigned short* __restrict__ Wbf,
             unsigned short* __restrict__ Y) {
    int id = blockIdx.x;
    int ct = id & 15;  id >>= 4;   // dout tile 0..15
    int mt = id & 31;  id >>= 5;   // m tile    0..31
    int r  = id % R_;
    int b  = id / R_;
    int t = threadIdx.x;
    int half = t >> 4;             // 0: lanes 0-15, 1: lanes 16-31
    int lane16 = t & 15;
    int m0 = mt * 16, c0 = ct * 16;

    v8f acc = {};
    const unsigned short* arow = nodebf + ((size_t)(b * N_ + m0 + lane16)) * DE_;
    const unsigned short* bcol = Wbf + ((size_t)(r * DOUT_ + c0 + lane16)) * DE_;

#pragma unroll
    for (int kc = 0; kc < 8; ++kc) {
        int k0 = kc * 32;
        ABf16 a, bm;
        // A 16x32 bf16 layout: lanes0-15 K={h8..h8+7, 16+h8..}, h=half
        a.q[0]  = *(const uint4*)(arow + k0 + half * 8);
        a.q[1]  = *(const uint4*)(arow + k0 + 16 + half * 8);
        // B 32x16 bf16 layout: same K sets as A ({h8..h8+7, 16+h8..}), col=lane16
        bm.q[0] = *(const uint4*)(bcol + k0 + half * 8);
        bm.q[1] = *(const uint4*)(bcol + k0 + 16 + half * 8);
        acc = __builtin_amdgcn_wmma_f32_16x16x32_bf16(
            false, a.v, false, bm.v, (short)0, acc, false, false);
    }

    // D layout: VGPR i -> row (i + half*8), col lane16. Store K-major: Y[..,c,m]
    uint4 pk;
    pk.x = (unsigned)f2bf(acc[0]) | ((unsigned)f2bf(acc[1]) << 16);
    pk.y = (unsigned)f2bf(acc[2]) | ((unsigned)f2bf(acc[3]) << 16);
    pk.z = (unsigned)f2bf(acc[4]) | ((unsigned)f2bf(acc[5]) << 16);
    pk.w = (unsigned)f2bf(acc[6]) | ((unsigned)f2bf(acc[7]) << 16);
    unsigned short* yp = Y + (((size_t)(b * R_ + r) * DOUT_) + c0 + lane16) * N_
                           + m0 + half * 8;
    *(uint4*)yp = pk;
}

// ---------------------------------------------------------------------------
// Kernel 2 (main): out[b,n0:16, c0:64] = sum_r adj[b,r] @ Y[b,r] + rowsum*S + bias
// One wave per (b, n-tile, 64-col group): 10*16*4 = 640 wmma per wave.
// adj converted f32->bf16 in registers; rowsum kept in f32.
// ---------------------------------------------------------------------------
__global__ __launch_bounds__(32)
void k_outgemm(const float* __restrict__ adj,
               const unsigned short* __restrict__ Y,
               const float* __restrict__ S,
               const float* __restrict__ bias,
               float* __restrict__ out) {
    int id = blockIdx.x;
    int cg = id & 3;  id >>= 2;    // column group (64 cols)
    int nt = id & 31; id >>= 5;    // n tile
    int b  = id;                   // batch
    int t = threadIdx.x;
    int half = t >> 4;
    int lane16 = t & 15;
    int n0 = nt * 16, c0 = cg * 64;

    v8f acc[4] = {};

    for (int r = 0; r < R_; ++r) {
        const float* arow = adj + (((size_t)(b * R_ + r)) * N_ + n0 + lane16) * N_;
        const unsigned short* ybase =
            Y + ((size_t)(b * R_ + r) * DOUT_ + c0 + lane16) * N_ + half * 8;
        float rsum = 0.f;

        for (int mc = 0; mc < 16; ++mc) {
            int m0 = mc * 32;
            // A operand: 16 adj f32 per lane (K sets per ISA A-layout), cvt to bf16
            float4 f0 = *(const float4*)(arow + m0 + half * 8);
            float4 f1 = *(const float4*)(arow + m0 + half * 8 + 4);
            float4 f2 = *(const float4*)(arow + m0 + 16 + half * 8);
            float4 f3 = *(const float4*)(arow + m0 + 16 + half * 8 + 4);
            rsum += f0.x + f0.y + f0.z + f0.w + f1.x + f1.y + f1.z + f1.w
                  + f2.x + f2.y + f2.z + f2.w + f3.x + f3.y + f3.z + f3.w;
            ABf16 a;
            a.s[0]  = f2bf(f0.x); a.s[1]  = f2bf(f0.y);
            a.s[2]  = f2bf(f0.z); a.s[3]  = f2bf(f0.w);
            a.s[4]  = f2bf(f1.x); a.s[5]  = f2bf(f1.y);
            a.s[6]  = f2bf(f1.z); a.s[7]  = f2bf(f1.w);
            a.s[8]  = f2bf(f2.x); a.s[9]  = f2bf(f2.y);
            a.s[10] = f2bf(f2.z); a.s[11] = f2bf(f2.w);
            a.s[12] = f2bf(f3.x); a.s[13] = f2bf(f3.y);
            a.s[14] = f2bf(f3.z); a.s[15] = f2bf(f3.w);
#pragma unroll
            for (int q = 0; q < 4; ++q) {
                const unsigned short* bp = ybase + (size_t)q * 16 * N_ + m0;
                ABf16 bm;
                bm.q[0] = *(const uint4*)bp;
                bm.q[1] = *(const uint4*)(bp + 16);
                acc[q] = __builtin_amdgcn_wmma_f32_16x16x32_bf16(
                    false, a.v, false, bm.v, (short)0, acc[q], false, false);
            }
        }

        // relation contribution: acc[q][i] += rowsum(row) * S[b,r,col]
        float sv0 = S[((b * R_ + r) * DOUT_) + c0 + 0  + lane16];
        float sv1 = S[((b * R_ + r) * DOUT_) + c0 + 16 + lane16];
        float sv2 = S[((b * R_ + r) * DOUT_) + c0 + 32 + lane16];
        float sv3 = S[((b * R_ + r) * DOUT_) + c0 + 48 + lane16];
#pragma unroll
        for (int i = 0; i < 8; ++i) {
            int row = i + half * 8;
            float rs = __shfl(rsum, row, 32) + __shfl(rsum, row + 16, 32);
            acc[0][i] += rs * sv0;
            acc[1][i] += rs * sv1;
            acc[2][i] += rs * sv2;
            acc[3][i] += rs * sv3;
        }
    }

    // epilogue: + bias, store f32
#pragma unroll
    for (int q = 0; q < 4; ++q) {
        float bb = bias[c0 + q * 16 + lane16];
#pragma unroll
        for (int i = 0; i < 8; ++i) {
            int row = n0 + i + half * 8;
            out[((size_t)b * N_ + row) * DOUT_ + c0 + q * 16 + lane16] = acc[q][i] + bb;
        }
    }
}

// ---------------------------------------------------------------------------
extern "C" void kernel_launch(void* const* d_in, const int* in_sizes, int n_in,
                              void* d_out, int out_size, void* d_ws, size_t ws_size,
                              hipStream_t stream) {
    const float* node = (const float*)d_in[0];   // [B,N,DE]
    const float* rel  = (const float*)d_in[1];   // [B,R,DR]
    const float* adj  = (const float*)d_in[2];   // [B,R,N,N]
    const float* W    = (const float*)d_in[3];   // [DOUT, INDIM]
    const float* bias = (const float*)d_in[4];   // [DOUT]
    float* out = (float*)d_out;                  // [B,N,DOUT]

    char* ws = (char*)d_ws;
    unsigned short* nodebf = (unsigned short*)(ws);                       //  4.00 MB
    unsigned short* Wbf    = (unsigned short*)(ws + 4u * 1024 * 1024);    //  1.25 MB
    float*          Smat   = (float*)(ws + 5505024u);                     //  0.16 MB
    unsigned short* Y      = (unsigned short*)(ws + 5668864u);            // 40.00 MB
    (void)in_sizes; (void)n_in; (void)out_size; (void)ws_size;

    // prep
    {
        int n = B_ * N_ * DE_;
        k_cvt_node<<<(n + 255) / 256, 256, 0, stream>>>(node, nodebf, n);
    }
    {
        int n = R_ * DOUT_ * DE_;
        k_cvt_w<<<(n + 255) / 256, 256, 0, stream>>>(W, Wbf);
    }
    {
        int n = B_ * R_ * DOUT_;
        k_srel<<<(n + 255) / 256, 256, 0, stream>>>(rel, W, Smat);
    }
    // Y = node @ Wn^T  (bf16 WMMA), K-major output
    {
        int blocks = B_ * R_ * (N_ / 16) * (DOUT_ / 16);   // 81920
        k_ygemm<<<blocks, 32, 0, stream>>>(nodebf, Wbf, Y);
    }
    // out = sum_r adj @ Y + rowsum*S + bias  (bf16 WMMA, f32 accumulate)
    {
        int blocks = B_ * (N_ / 16) * (DOUT_ / 64);        // 2048
        k_outgemm<<<blocks, 32, 0, stream>>>(adj, Y, Smat, bias, out);
    }
}
